// VariationalLEncoder_7765300871250
// MI455X (gfx1250) — matrix operand order repaired and checked by reading, not compiled, not run on hardware
//
#include <hip/hip_runtime.h>
#include <hip/hip_bf16.h>

// ---------------------------------------------------------------------------
// Variational GCN encoder for MI455X (gfx1250), wave32 + WMMA bf16.
//
// Roofline: GEMMs ~11 GFLOP (negligible); edge gather/scatter ~3.3 GB of
// random 512B-row traffic. hp[N,128] (51.2MB) and agg[N,128] (51.2MB) both
// fit in the 192MB L2, so scatter runs at L2 bandwidth with native
// global_atomic_add_f32. GEMMs use v_wmma_f32_16x16x32_bf16 (f32 accum).
// bf16 conversion uses hardware converts (cast to __bf16), not manual RNE.
// ---------------------------------------------------------------------------

typedef __attribute__((ext_vector_type(16))) __bf16 v16bf;
typedef __attribute__((ext_vector_type(8)))  float  v8f;

union ABfrag {
    v16bf v;
    __bf16 h[16];
    uint4 q[2];
};

__device__ __forceinline__ void atomicAddF(float* p, float v) {
    __hip_atomic_fetch_add(p, v, __ATOMIC_RELAXED, __HIP_MEMORY_SCOPE_AGENT);
}

// --------------------------- weight conversion -----------------------------
// W1p  [256][32] bf16 : W1[n][k] padded K20->32 (B fragment for layer 1)
// WgT  [128][256] bf16: Wg transposed (B fragment column n contiguous in K)
// Wmub/Wlsb [128][64] bf16: row n = column n of B (Wmu.T GEMM)
__global__ void convert_weights(const float* __restrict__ W1,
                                const float* __restrict__ Wg,
                                const float* __restrict__ Wmu,
                                const float* __restrict__ Wls,
                                __bf16* __restrict__ W1p,
                                __bf16* __restrict__ WgT,
                                __bf16* __restrict__ Wmub,
                                __bf16* __restrict__ Wlsb) {
    int i = blockIdx.x * blockDim.x + threadIdx.x;
    if (i < 256 * 32) {
        int n = i >> 5, k = i & 31;
        W1p[i] = (k < 20) ? (__bf16)W1[n * 20 + k] : (__bf16)0.f;
    }
    if (i < 128 * 256) {
        int n = i >> 8, k = i & 255;
        WgT[i] = (__bf16)Wg[k * 128 + n];
    }
    if (i < 128 * 64) {
        Wmub[i] = (__bf16)Wmu[i];
        Wlsb[i] = (__bf16)Wls[i];
    }
}

// ------------------- fused layer1 + GCN projection (WMMA) ------------------
// One wave handles a 16-row tile:
//   stage 1: h16 = relu(x_tile @ W1.T + b1)   [16,256] -> LDS bf16
//   stage 2: hp_tile = h16 @ Wg               [16,128] -> global f32
__global__ void __launch_bounds__(128)
fused_l1_proj(const float* __restrict__ x,
              const __bf16* __restrict__ W1p,
              const float* __restrict__ b1,
              const __bf16* __restrict__ WgT,
              float* __restrict__ hp, int N) {
    __shared__ __bf16 lds[4][16 * 256];           // 32 KB: bf16 h16 per wave

    const int wave = threadIdx.x >> 5;
    const int lane = threadIdx.x & 31;
    const int tile = blockIdx.x * 4 + wave;
    if (tile * 16 >= N) return;                   // wave-uniform exit

    const int r  = lane & 15;                     // row (A) / col (B,C)
    const int hi = lane >> 4;                     // lane-half select
    const int row = tile * 16 + r;

    // A fragment from x[N,20], K padded to 32.
    // lanes 0-15: K {0..7, 16..23}; lanes 16-31: K {8..15, 24..31}
    ABfrag a;
#pragma unroll
    for (int i = 0; i < 16; ++i) {
        int k = i + (i >= 8 ? 8 : 0) + hi * 8;
        a.h[i] = (k < 20) ? (__bf16)x[row * 20 + k] : (__bf16)0.f;
    }

    __bf16* myl = lds[wave];

    // -------- stage 1: 16 n-tiles of 16 cols each (256 outputs) --------
    for (int nt = 0; nt < 16; ++nt) {
        // B fragment: col n = nt*16+r; lanes 0-15 K {0..7, 16..23}, lanes 16-31 K {8..15, 24..31}
        const uint4* pb = (const uint4*)(W1p + ((nt * 16 + r) * 32 + hi * 8));
        ABfrag b; b.q[0] = pb[0]; b.q[1] = pb[2];
        v8f c = {};
        c = __builtin_amdgcn_wmma_f32_16x16x32_bf16(false, a.v, false, b.v,
                                                    (short)0, c, false, false);
        // C layout: N = r, M = vr + hi*8
        int col = nt * 16 + r;
        float bias = b1[col];
#pragma unroll
        for (int vr = 0; vr < 8; ++vr) {
            int m   = vr + hi * 8;
            float v = c[vr] + bias;
            v = v > 0.f ? v : 0.f;
            myl[m * 256 + col] = (__bf16)v;
        }
    }
    // Same-wave LDS ops are in-order; no cross-wave sharing -> no barrier.

    // -------- stage 2: hp_tile[16,128] = h16[16,256] @ Wg[256,128] --------
    for (int nt = 0; nt < 8; ++nt) {
        v8f c = {};
        for (int kc = 0; kc < 8; ++kc) {
            // A from LDS: row r, K groups {kc*32+hi*8 ..+7} and {+16..+23}
            const uint4* pa = (const uint4*)(myl + r * 256 + kc * 32 + hi * 8);
            ABfrag a2; a2.q[0] = pa[0]; a2.q[1] = pa[2];
            // B from WgT: col n = nt*16+r, K groups {kc*32+hi*8 ..+7} and {+16..+23}
            const uint4* pb = (const uint4*)(WgT + (nt * 16 + r) * 256 + kc * 32 + hi * 8);
            ABfrag b2; b2.q[0] = pb[0]; b2.q[1] = pb[2];
            c = __builtin_amdgcn_wmma_f32_16x16x32_bf16(false, a2.v, false, b2.v,
                                                        (short)0, c, false, false);
        }
#pragma unroll
        for (int vr = 0; vr < 8; ++vr) {
            int m = vr + hi * 8;
            hp[(size_t)(tile * 16 + m) * 128 + nt * 16 + r] = c[vr];
        }
    }
}

// ----------------------------- graph kernels -------------------------------
__global__ void zero_f32(float* __restrict__ p, int n) {
    int i = blockIdx.x * blockDim.x + threadIdx.x;
    if (i < n) p[i] = 0.f;
}

__global__ void degree_count(const long long* __restrict__ ei, float* __restrict__ deg,
                             int E) {
    int i = blockIdx.x * blockDim.x + threadIdx.x;
    int stride = gridDim.x * blockDim.x;
    for (int e = i; e < E; e += stride) {
        int d = (int)ei[(size_t)E + e];           // dst row
        atomicAddF(&deg[d], 1.0f);
    }
}

__global__ void make_dis(const float* __restrict__ deg, float* __restrict__ dis, int N) {
    int i = blockIdx.x * blockDim.x + threadIdx.x;
    if (i < N) dis[i] = rsqrtf(deg[i] + 1.0f);    // +1 = self loop; deg>=1 always
}

// agg[i] = dis(n)^2 * hp[i]  (self-loop contribution; fully initializes agg)
__global__ void agg_init(const float* __restrict__ hp, const float* __restrict__ dis,
                         float* __restrict__ agg, int N) {
    int i = blockIdx.x * blockDim.x + threadIdx.x;
    int total = N * 128;
    if (i < total) {
        int n = i >> 7;
        float d = dis[n];
        agg[i] = d * d * hp[i];
    }
}

// One wave per edge: lane handles 4 contiguous channels (float4 gather from
// L2-resident hp, 4 native f32 atomic adds into L2-resident agg).
__global__ void edge_scatter(const long long* __restrict__ ei,
                             const float* __restrict__ dis,
                             const float* __restrict__ hp,
                             float* __restrict__ agg, int E) {
    int wid    = (blockIdx.x * blockDim.x + threadIdx.x) >> 5;
    int lane   = threadIdx.x & 31;
    int stride = (gridDim.x * blockDim.x) >> 5;
    for (int e = wid; e < E; e += stride) {
        int s = (int)ei[e];
        int d = (int)ei[(size_t)E + e];
        float nrm = dis[s] * dis[d];
        float4 v = *(const float4*)(hp + (size_t)s * 128 + lane * 4);
        float* o = agg + (size_t)d * 128 + lane * 4;
        atomicAddF(o + 0, nrm * v.x);
        atomicAddF(o + 1, nrm * v.y);
        atomicAddF(o + 2, nrm * v.z);
        atomicAddF(o + 3, nrm * v.w);
    }
}

// ----------------- finalize GCN + variational heads (WMMA) -----------------
// h2 = relu(agg + bg); mu = h2[:, :64], ls = h2[:, 64:]
// out[0:N*128]      = mu @ Wmu.T + bmu
// out[N*128:2N*128] = ls @ Wls.T + bls
__global__ void __launch_bounds__(128)
finalize_heads(const float* __restrict__ agg, const float* __restrict__ bg,
               const __bf16* __restrict__ Wmub, const float* __restrict__ bmu,
               const __bf16* __restrict__ Wlsb, const float* __restrict__ bls,
               float* __restrict__ out, int N) {
    const int wave = threadIdx.x >> 5;
    const int lane = threadIdx.x & 31;
    const int tile = blockIdx.x * 4 + wave;
    if (tile * 16 >= N) return;                   // wave-uniform exit

    const int r  = lane & 15;
    const int hi = lane >> 4;
    const int row = tile * 16 + r;
    const float* arow = agg + (size_t)row * 128;

    for (int head = 0; head < 2; ++head) {
        const __bf16* Wb = head ? Wlsb : Wmub;
        const float* bias = head ? bls : bmu;
        float* o = out + (size_t)head * N * 128;

        // Two K=32 A fragments covering this head's 64 channels.
        // Per chunk, lane needs two contiguous 8-float groups:
        //   group0 at k = kc*32 + hi*8, group1 at k = kc*32 + 16 + hi*8
        ABfrag a[2];
#pragma unroll
        for (int kc = 0; kc < 2; ++kc) {
#pragma unroll
            for (int g = 0; g < 2; ++g) {
                int base = head * 64 + kc * 32 + g * 16 + hi * 8;
                float4 v0 = *(const float4*)(arow + base);
                float4 v1 = *(const float4*)(arow + base + 4);
                float t[8] = { v0.x + bg[base + 0], v0.y + bg[base + 1],
                               v0.z + bg[base + 2], v0.w + bg[base + 3],
                               v1.x + bg[base + 4], v1.y + bg[base + 5],
                               v1.z + bg[base + 6], v1.w + bg[base + 7] };
#pragma unroll
                for (int j = 0; j < 8; ++j) {
                    float v = t[j] > 0.f ? t[j] : 0.f;
                    a[kc].h[g * 8 + j] = (__bf16)v;
                }
            }
        }

        for (int nt = 0; nt < 8; ++nt) {
            v8f c = {};
#pragma unroll
            for (int kc = 0; kc < 2; ++kc) {
                const uint4* pb = (const uint4*)(Wb + (nt * 16 + r) * 64 + kc * 32 + hi * 8);
                ABfrag b; b.q[0] = pb[0]; b.q[1] = pb[2];
                c = __builtin_amdgcn_wmma_f32_16x16x32_bf16(false, a[kc].v, false, b.v,
                                                            (short)0, c, false, false);
            }
            int col = nt * 16 + r;
            float bo = bias[col];
#pragma unroll
            for (int vr = 0; vr < 8; ++vr) {
                int m = vr + hi * 8;
                o[(size_t)(tile * 16 + m) * 128 + col] = c[vr] + bo;
            }
        }
    }
}

// ------------------------------- launcher ----------------------------------
extern "C" void kernel_launch(void* const* d_in, const int* in_sizes, int n_in,
                              void* d_out, int out_size, void* d_ws, size_t ws_size,
                              hipStream_t stream) {
    const float*     x   = (const float*)d_in[0];
    const long long* ei  = (const long long*)d_in[1];
    const float*     W1  = (const float*)d_in[2];
    const float*     b1  = (const float*)d_in[3];
    const float*     Wg  = (const float*)d_in[4];
    const float*     bg  = (const float*)d_in[5];
    const float*     Wmu = (const float*)d_in[6];
    const float*     bmu = (const float*)d_in[7];
    const float*     Wls = (const float*)d_in[8];
    const float*     bls = (const float*)d_in[9];

    const int N = in_sizes[0] / 20;
    const int E = in_sizes[1] / 2;

    // --- workspace carve (256B aligned) ---
    char* ws = (char*)d_ws;
    size_t off = 0;
    auto carve = [&](size_t bytes) -> void* {
        void* p = ws + off;
        off += (bytes + 255) & ~(size_t)255;
        return p;
    };
    float* hp   = (float*)carve((size_t)N * 128 * sizeof(float));  // 51.2 MB
    float* agg  = (float*)carve((size_t)N * 128 * sizeof(float));  // 51.2 MB
    float* deg  = (float*)carve((size_t)N * sizeof(float));
    float* dis  = (float*)carve((size_t)N * sizeof(float));
    __bf16* W1p  = (__bf16*)carve(256 * 32 * 2);
    __bf16* WgT  = (__bf16*)carve(128 * 256 * 2);
    __bf16* Wmub = (__bf16*)carve(128 * 64 * 2);
    __bf16* Wlsb = (__bf16*)carve(128 * 64 * 2);

    const int tiles  = (N + 15) / 16;
    const int tblks  = (tiles + 3) / 4;

    convert_weights<<<(128 * 256 + 255) / 256, 256, 0, stream>>>(
        W1, Wg, Wmu, Wls, W1p, WgT, Wmub, Wlsb);

    fused_l1_proj<<<tblks, 128, 0, stream>>>(x, W1p, b1, WgT, hp, N);

    zero_f32<<<(N + 255) / 256, 256, 0, stream>>>(deg, N);
    degree_count<<<2048, 256, 0, stream>>>(ei, deg, E);
    make_dis<<<(N + 255) / 256, 256, 0, stream>>>(deg, dis, N);

    agg_init<<<(N * 128 + 255) / 256, 256, 0, stream>>>(hp, dis, agg, N);
    edge_scatter<<<16384, 256, 0, stream>>>(ei, dis, hp, agg, E);

    finalize_heads<<<tblks, 128, 0, stream>>>(agg, bg, Wmub, bmu, Wlsb, bls,
                                              (float*)d_out, N);
}
